// MambaBlock_23381801959698
// MI455X (gfx1250) — matrix operand order repaired and checked by reading, not compiled, not run on hardware
//
#include <hip/hip_runtime.h>
#include <hip/hip_bf16.h>
#include <stdint.h>

// ---------------------------------------------------------------------------
// MI455X (gfx1250) Mamba block, compile-only target.
// Strategy: bf16 WMMA (v_wmma_f32_16x16x32_bf16) for all 6 GEMMs (246 GFLOP),
// CDNA5 async global->LDS DMA (global_load_async_to_lds_b128 / ASYNCcnt) for
// GEMM operand staging, fp32 VALU for rmsnorm/conv/softplus/selective-scan.
// Reference dims: B=2, L=4096, d_model=1024, d_inner=2048, d_state=16,
// d_conv=4, dt_rank=64.
//
// GEMM invariants relied on below: M == 8192 (multiple of 128); every N
// (4096, 96, 2048, 1024) is a multiple of 16; every K is a multiple of 32.
// ---------------------------------------------------------------------------

typedef __attribute__((ext_vector_type(16))) __bf16 v16bf;
typedef __attribute__((ext_vector_type(8)))  __bf16 v8bf;
typedef __attribute__((ext_vector_type(8)))  float  v8f;

#define RMS_EPS 1.1920929e-07f

__device__ __forceinline__ float sigmoidf_(float x) { return 1.f / (1.f + __expf(-x)); }

// CDNA5 async global->LDS copy, 16B per lane. LDS destination is the low 32
// bits of the generic pointer (LDS aperture: addr[31:0] == LDS byte offset).
// Tracked by ASYNCcnt; retire with s_wait_asynccnt before the barrier.
__device__ __forceinline__ void async_ldg_b128(const void* lds_ptr, const void* gptr) {
    uint32_t lo = (uint32_t)(uintptr_t)lds_ptr;
    uint64_t ga = (uint64_t)(uintptr_t)gptr;
    asm volatile("global_load_async_to_lds_b128 %0, %1, off"
                 :: "v"(lo), "v"(ga) : "memory");
}
__device__ __forceinline__ void wait_asynccnt0() {
    asm volatile("s_wait_asynccnt 0x0" ::: "memory");
}

// ---------------------------------------------------------------------------
// float -> bf16 convert (weights, activations at GEMM boundaries)
// ---------------------------------------------------------------------------
__global__ void f2bf_kernel(const float* __restrict__ in, __bf16* __restrict__ out, long n) {
    long i = (long)blockIdx.x * blockDim.x + threadIdx.x;
    if (i < n) out[i] = (__bf16)in[i];
}

// ---------------------------------------------------------------------------
// rmsnorm over 1024 columns, output bf16. One block (256 thr, 8 waves) / row.
// ---------------------------------------------------------------------------
__global__ __launch_bounds__(256) void rmsnorm_bf16_kernel(
    const float* __restrict__ x, __bf16* __restrict__ out, int ncols) {
    const int row = blockIdx.x;
    const float* xr = x + (size_t)row * ncols;
    float v[4];
    float s = 0.f;
#pragma unroll
    for (int i = 0; i < 4; ++i) {
        int c = threadIdx.x + i * 256;
        v[i] = (c < ncols) ? xr[c] : 0.f;
        s += v[i] * v[i];
    }
#pragma unroll
    for (int off = 16; off > 0; off >>= 1) s += __shfl_xor(s, off, 32);
    __shared__ float red[8];
    if ((threadIdx.x & 31) == 0) red[threadIdx.x >> 5] = s;
    __syncthreads();
    float tot = 0.f;
#pragma unroll
    for (int i = 0; i < 8; ++i) tot += red[i];
    const float scale = rsqrtf(tot / (float)ncols + RMS_EPS);
#pragma unroll
    for (int i = 0; i < 4; ++i) {
        int c = threadIdx.x + i * 256;
        if (c < ncols) out[(size_t)row * ncols + c] = (__bf16)(v[i] * scale);
    }
}

// residual add + rmsnorm: x1 = x + res ; xn = bf16(rmsnorm(x1))
__global__ __launch_bounds__(256) void add_rmsnorm_bf16_kernel(
    const float* __restrict__ x, const float* __restrict__ res,
    float* __restrict__ x1, __bf16* __restrict__ xn, int ncols) {
    const int row = blockIdx.x;
    const size_t base = (size_t)row * ncols;
    float v[4];
    float s = 0.f;
#pragma unroll
    for (int i = 0; i < 4; ++i) {
        int c = threadIdx.x + i * 256;
        float t = (c < ncols) ? (x[base + c] + res[base + c]) : 0.f;
        v[i] = t;
        s += t * t;
        if (c < ncols) x1[base + c] = t;
    }
#pragma unroll
    for (int off = 16; off > 0; off >>= 1) s += __shfl_xor(s, off, 32);
    __shared__ float red[8];
    if ((threadIdx.x & 31) == 0) red[threadIdx.x >> 5] = s;
    __syncthreads();
    float tot = 0.f;
#pragma unroll
    for (int i = 0; i < 8; ++i) tot += red[i];
    const float scale = rsqrtf(tot / (float)ncols + RMS_EPS);
#pragma unroll
    for (int i = 0; i < 4; ++i) {
        int c = threadIdx.x + i * 256;
        if (c < ncols) xn[base + c] = (__bf16)(v[i] * scale);
    }
}

// ---------------------------------------------------------------------------
// bf16 WMMA GEMM:  C[M,N] (f32) = A[M,K] (bf16, row-major) x W[N,K]^T (bf16)
// 128x128 block tile, K-tile 32, 256 threads = 8 waves (4 over M x 2 over N),
// each wave: 2x4 = 8 v_wmma_f32_16x16x32_bf16 per K-step. Double-buffered LDS
// with [128][40] bf16 rows (80B stride -> conflict-free 16-lane ds_load_b128).
// Operand staging uses CDNA5 async global->LDS DMA (no VGPR round trip):
// per K-step: issue 4 async b128/thread for tile kt+1, compute tile kt,
// s_wait_asynccnt 0, s_barrier.
//
// OOB note (N=96 GEMM only): W rows N..127 are loaded as garbage but each C
// column depends on exactly one W row, and the epilogue skips n-tiles beyond
// N (N is a multiple of 16, so the check is uniform per tile). The stray reads
// stay inside the workspace (weight buffers sit at its front). M is always a
// multiple of 128 -> no row guards anywhere.
// ---------------------------------------------------------------------------
__global__ __launch_bounds__(256) void gemm_bf16_wmma_kernel(
    const __bf16* __restrict__ A, const __bf16* __restrict__ W,
    float* __restrict__ C, int M, int N, int K) {
    __shared__ __bf16 As[2][128][40];
    __shared__ __bf16 Bs[2][128][40];

    const int tid    = threadIdx.x;
    const int lane   = tid & 31;
    const int wid    = tid >> 5;
    const int blockM = blockIdx.y * 128;
    const int blockN = blockIdx.x * 128;
    const int wm     = (wid & 3) * 32;   // wave M offset (4 waves x 32 rows)
    const int wn     = (wid >> 2) * 64;  // wave N offset (2 waves x 64 cols)
    const int ldRow  = tid >> 1;         // 128 rows, 2 threads/row
    const int ldCol  = (tid & 1) * 16;   // 16 bf16 (32B) per thread
    const int nK     = K >> 5;           // all K are multiples of 32

    const __bf16* aRow = A + (size_t)(blockM + ldRow) * K + ldCol;
    const __bf16* wRow = W + (size_t)(blockN + ldRow) * K + ldCol;

    v8f acc[2][4] = {};

    auto issue_async = [&](int kt, int buf) {
        const int kb = kt * 32;
        async_ldg_b128(&As[buf][ldRow][ldCol],     aRow + kb);
        async_ldg_b128(&As[buf][ldRow][ldCol + 8], aRow + kb + 8);
        async_ldg_b128(&Bs[buf][ldRow][ldCol],     wRow + kb);
        async_ldg_b128(&Bs[buf][ldRow][ldCol + 8], wRow + kb + 8);
    };

    issue_async(0, 0);
    wait_asynccnt0();
    __syncthreads();

    const int fm = lane & 15;   // row (A: M, B: N) within 16x16 tile
    const int kg = lane >> 4;   // K half selector per ISA 16-bit layouts

    for (int kt = 0; kt < nK; ++kt) {
        const int buf = kt & 1;
        if (kt + 1 < nK) {
            issue_async(kt + 1, buf ^ 1);     // DMA overlaps compute below
            if (kt + 2 < nK) {                // speculative L2 prefetch 2 ahead
                __builtin_prefetch(aRow + (kt + 2) * 32, 0, 1);
                __builtin_prefetch(wRow + (kt + 2) * 32, 0, 1);
            }
        }

        // A fragments: lane m=fm; VGPR0-3 hold K = kg*8..kg*8+7, VGPR4-7 hold +16
        v16bf afrag[2];
#pragma unroll
        for (int i = 0; i < 2; ++i) {
            const __bf16* p = &As[buf][wm + i * 16 + fm][kg * 8];
            v8bf lo = *(const v8bf*)p;
            v8bf hi = *(const v8bf*)(p + 16);
            afrag[i] = __builtin_shufflevector(lo, hi, 0,1,2,3,4,5,6,7,8,9,10,11,12,13,14,15);
        }
        // B fragments: lane n=fm; VGPR0-3 hold K = kg*8..kg*8+7, VGPR4-7 hold +16
        v16bf bfrag[4];
#pragma unroll
        for (int j = 0; j < 4; ++j) {
            const __bf16* p = &Bs[buf][wn + j * 16 + fm][kg * 8];
            v8bf lo = *(const v8bf*)p;
            v8bf hi = *(const v8bf*)(p + 16);
            bfrag[j] = __builtin_shufflevector(lo, hi, 0,1,2,3,4,5,6,7,8,9,10,11,12,13,14,15);
        }
#pragma unroll
        for (int i = 0; i < 2; ++i)
#pragma unroll
            for (int j = 0; j < 4; ++j)
                acc[i][j] = __builtin_amdgcn_wmma_f32_16x16x32_bf16(
                    false, afrag[i], false, bfrag[j], (short)0, acc[i][j], false, false);

        if (kt + 1 < nK) wait_asynccnt0();    // retire DMA for tile kt+1
        __syncthreads();
    }

    // Epilogue. C/D layout: lanes 0-15 hold M=vgpr (0..7), lanes 16-31 M=8+vgpr.
    // Per store instruction the wave writes two coalesced 64B row segments
    // (lanes 0-15: cols col..col+15 of row r; lanes 16-31: same of row r+8).
    // n-tile guard is scalar (N % 16 == 0); no row guards (M % 128 == 0).
#pragma unroll
    for (int i = 0; i < 2; ++i) {
#pragma unroll
        for (int j = 0; j < 4; ++j) {
            const int ntile = blockN + wn + j * 16;
            if (ntile + 16 <= N) {
                const int row0 = blockM + wm + i * 16 + kg * 8;
                float* p = C + (size_t)row0 * N + (ntile + fm);
#pragma unroll
                for (int v = 0; v < 8; ++v) {
                    *p = acc[i][j][v];
                    p += N;
                }
            }
        }
    }
}

// ---------------------------------------------------------------------------
// Depthwise causal conv (d_conv=4) + bias + SiLU. xp = xz[..., :d_inner].
// Outputs both f32 (for scan / D path) and bf16 (for x_proj GEMM).
// ---------------------------------------------------------------------------
__global__ __launch_bounds__(256) void conv_silu_kernel(
    const float* __restrict__ xz, const float* __restrict__ cw,
    const float* __restrict__ cb, float* __restrict__ xcf,
    __bf16* __restrict__ xcb, int L, int DI, int XZ, long total) {
    long idx = (long)blockIdx.x * 256 + threadIdx.x;
    if (idx >= total) return;
    const int d  = (int)(idx % DI);
    const long bl = idx / DI;
    const int l  = (int)(bl % L);
    const long brow = bl - l;  // b*L
    float acc = cb[d];
#pragma unroll
    for (int k = 0; k < 4; ++k) {
        const int ls = l + k - 3;
        if (ls >= 0) acc += xz[(brow + ls) * (long)XZ + d] * cw[d * 4 + k];
    }
    const float sv = acc * sigmoidf_(acc);
    xcf[idx] = sv;
    xcb[idx] = (__bf16)sv;
}

// x_dbl[:, :dt_rank] -> bf16 (dt_proj GEMM input)
__global__ void extract_dt_kernel(const float* __restrict__ xdbl,
                                  __bf16* __restrict__ dtin,
                                  int XD, int DTR, long total) {
    long i = (long)blockIdx.x * 256 + threadIdx.x;
    if (i >= total) return;
    const long r = i / DTR;
    const int  c = (int)(i % DTR);
    dtin[i] = (__bf16)xdbl[r * XD + c];
}

// ---------------------------------------------------------------------------
// Selective scan. One 16-lane group per (b, d): lane = state index s.
// h kept in register, dt-softplus fused, y reduced over states with a
// 4-step shfl_xor butterfly (stays inside the 16-lane half of the wave32).
// ---------------------------------------------------------------------------
__global__ __launch_bounds__(256) void scan_kernel(
    const float* __restrict__ dtraw, const float* __restrict__ dtb,
    const float* __restrict__ xcf,   const float* __restrict__ xdbl,
    const float* __restrict__ A_log, float* __restrict__ y,
    int Bsz, int L, int DI, int DS, int XD, int DTR) {
    const int tid = blockIdx.x * 256 + threadIdx.x;
    const int s = tid & 15;
    const int g = tid >> 4;
    if (g >= Bsz * DI) return;
    const int d = g % DI;
    const int b = g / DI;

    const float A    = -__expf(A_log[d * DS + s]);
    const float bias = dtb[d];
    float h = 0.f;

    for (int t = 0; t < L; ++t) {
        const long rb = (long)b * L + t;
        const float dtv = dtraw[rb * DI + d] + bias;
        const float dt  = (dtv > 20.f) ? dtv : log1pf(__expf(dtv));  // softplus
        const float u   = xcf[rb * DI + d];
        const float Bt  = xdbl[rb * XD + DTR + s];
        const float Ct  = xdbl[rb * XD + DTR + DS + s];
        h = h * __expf(dt * A) + (dt * u) * Bt;
        float c = h * Ct;
        c += __shfl_xor(c, 8, 32);
        c += __shfl_xor(c, 4, 32);
        c += __shfl_xor(c, 2, 32);
        c += __shfl_xor(c, 1, 32);
        if (s == 0) y[rb * DI + d] = c;
    }
}

// y2 = (y + xc*D) * silu(z), z = xz[..., d_inner:] ; output bf16 for out_proj
__global__ void gate_kernel(const float* __restrict__ y, const float* __restrict__ xcf,
                            const float* __restrict__ Dp, const float* __restrict__ xz,
                            __bf16* __restrict__ y2, int DI, int XZ, long total) {
    long i = (long)blockIdx.x * 256 + threadIdx.x;
    if (i >= total) return;
    const int d = (int)(i % DI);
    const long r = i / DI;
    const float yv = y[i] + xcf[i] * Dp[d];
    const float zv = xz[r * (long)XZ + DI + d];
    y2[i] = (__bf16)(yv * (zv * sigmoidf_(zv)));
}

__global__ void silu_bf16_kernel(const float* __restrict__ in, __bf16* __restrict__ out, long n) {
    long i = (long)blockIdx.x * 256 + threadIdx.x;
    if (i < n) {
        const float v = in[i];
        out[i] = (__bf16)(v * sigmoidf_(v));
    }
}

__global__ void add_kernel(const float* __restrict__ a, const float* __restrict__ b,
                           float* __restrict__ o, long n) {
    long i = (long)blockIdx.x * 256 + threadIdx.x;
    if (i < n) o[i] = a[i] + b[i];
}

// ---------------------------------------------------------------------------
extern "C" void kernel_launch(void* const* d_in, const int* in_sizes, int n_in,
                              void* d_out, int out_size, void* d_ws, size_t ws_size,
                              hipStream_t stream) {
    (void)in_sizes; (void)n_in; (void)out_size; (void)ws_size;

    constexpr int Bsz = 2, L = 4096, DM = 1024, DI = 2048, DS = 16, DTR = 64, XD = 96;
    constexpr int M  = Bsz * L;   // 8192 rows
    constexpr int N2 = 2 * DI;    // 4096 (in_proj out)
    constexpr int H  = 4 * DM;    // 4096 (mlp hidden)

    const float* x         = (const float*)d_in[0];
    const float* in_proj_w = (const float*)d_in[1];
    const float* conv_w    = (const float*)d_in[2];
    const float* conv_b    = (const float*)d_in[3];
    const float* x_proj_w  = (const float*)d_in[4];
    const float* dt_proj_w = (const float*)d_in[5];
    const float* dt_proj_b = (const float*)d_in[6];
    const float* A_log     = (const float*)d_in[7];
    const float* Dp        = (const float*)d_in[8];
    const float* out_proj_w= (const float*)d_in[9];
    const float* mlp_w1    = (const float*)d_in[10];
    const float* mlp_w2    = (const float*)d_in[11];
    float* out = (float*)d_out;

    // ---- workspace bump allocator (256B aligned) ----
    char* base = (char*)d_ws;
    size_t off = 0;
    auto alloc = [&](size_t bytes) -> char* {
        char* p = base + off;
        off = (off + bytes + 255) & ~(size_t)255;
        return p;
    };
    __bf16* wbf_in  = (__bf16*)alloc((size_t)N2 * DM * 2);
    __bf16* wbf_xp  = (__bf16*)alloc((size_t)XD * DI * 2);
    __bf16* wbf_dt  = (__bf16*)alloc((size_t)DI * DTR * 2);
    __bf16* wbf_out = (__bf16*)alloc((size_t)DM * DI * 2);
    __bf16* wbf_w1  = (__bf16*)alloc((size_t)H * DM * 2);
    __bf16* wbf_w2  = (__bf16*)alloc((size_t)DM * H * 2);
    __bf16* xn_bf   = (__bf16*)alloc((size_t)M * DM * 2);
    float*  xz_f    = (float*) alloc((size_t)M * N2 * 4);  // reused as h1_f
    float*  xc_f    = (float*) alloc((size_t)M * DI * 4);
    __bf16* xc_bf   = (__bf16*)alloc((size_t)M * DI * 2);
    float*  xdbl_f  = (float*) alloc((size_t)M * XD * 4);
    __bf16* dtin_bf = (__bf16*)alloc((size_t)M * DTR * 2);
    float*  dtraw_f = (float*) alloc((size_t)M * DI * 4);
    float*  y_f     = (float*) alloc((size_t)M * DI * 4);
    __bf16* y2_bf   = (__bf16*)alloc((size_t)M * DI * 2);
    float*  mo_f    = (float*) alloc((size_t)M * DM * 4);  // mamba out, reused as mlp out
    float*  x1_f    = (float*) alloc((size_t)M * DM * 4);
    __bf16* xn2_bf  = (__bf16*)alloc((size_t)M * DM * 2);
    __bf16* h1_bf   = (__bf16*)alloc((size_t)M * H * 2);
    float*  h1_f    = xz_f;  // alias: xz dead after gate_kernel
    float*  mlp_f   = mo_f;  // alias: mamba_out dead after add_rmsnorm

    auto cvt = [&](const float* src, __bf16* dst, long n) {
        f2bf_kernel<<<(unsigned)((n + 255) / 256), 256, 0, stream>>>(src, dst, n);
    };
    // weight conversions (cheap, ~30MB total)
    cvt(in_proj_w,  wbf_in,  (long)N2 * DM);
    cvt(x_proj_w,   wbf_xp,  (long)XD * DI);
    cvt(dt_proj_w,  wbf_dt,  (long)DI * DTR);
    cvt(out_proj_w, wbf_out, (long)DM * DI);
    cvt(mlp_w1,     wbf_w1,  (long)H * DM);
    cvt(mlp_w2,     wbf_w2,  (long)DM * H);

    const dim3 blk(256);
    auto gemm = [&](const __bf16* A, const __bf16* W, float* C, int m, int n, int k) {
        dim3 grid((n + 127) / 128, (m + 127) / 128);
        gemm_bf16_wmma_kernel<<<grid, blk, 0, stream>>>(A, W, C, m, n, k);
    };

    // 1) xn = rmsnorm(x)
    rmsnorm_bf16_kernel<<<M, blk, 0, stream>>>(x, xn_bf, DM);
    // 2) xz = xn @ in_proj^T            [8192 x 4096]
    gemm(xn_bf, wbf_in, xz_f, M, N2, DM);
    // 3) xc = silu(causal_conv(xp) + b) [8192 x 2048]
    conv_silu_kernel<<<(M * DI) / 256, blk, 0, stream>>>(xz_f, conv_w, conv_b,
                                                         xc_f, xc_bf, L, DI, N2, (long)M * DI);
    // 4) x_dbl = xc @ x_proj^T          [8192 x 96]
    gemm(xc_bf, wbf_xp, xdbl_f, M, XD, DI);
    // 5) dt input slice -> bf16
    extract_dt_kernel<<<(M * DTR) / 256, blk, 0, stream>>>(xdbl_f, dtin_bf, XD, DTR, (long)M * DTR);
    // 6) dt_raw = dt_in @ dt_proj^T     [8192 x 2048]
    gemm(dtin_bf, wbf_dt, dtraw_f, M, DI, DTR);
    // 7) selective scan (softplus fused)
    scan_kernel<<<(Bsz * DI * DS) / 256, blk, 0, stream>>>(dtraw_f, dt_proj_b, xc_f, xdbl_f,
                                                           A_log, y_f, Bsz, L, DI, DS, XD, DTR);
    // 8) gate: y2 = (y + xc*D) * silu(z)
    gate_kernel<<<(M * DI) / 256, blk, 0, stream>>>(y_f, xc_f, Dp, xz_f, y2_bf,
                                                    DI, N2, (long)M * DI);
    // 9) mamba_out = y2 @ out_proj^T    [8192 x 1024]
    gemm(y2_bf, wbf_out, mo_f, M, DM, DI);
    // 10) x1 = x + mamba_out ; xn2 = rmsnorm(x1)
    add_rmsnorm_bf16_kernel<<<M, blk, 0, stream>>>(x, mo_f, x1_f, xn2_bf, DM);
    // 11) h1 = xn2 @ w1^T               [8192 x 4096]
    gemm(xn2_bf, wbf_w1, h1_f, M, H, DM);
    // 12) h1 = silu(h1) -> bf16
    silu_bf16_kernel<<<(M * H) / 256, blk, 0, stream>>>(h1_f, h1_bf, (long)M * H);
    // 13) mlp_out = h1 @ w2^T           [8192 x 1024]
    gemm(h1_bf, wbf_w2, mlp_f, M, DM, H);
    // 14) out = x1 + mlp_out
    add_kernel<<<(M * DM) / 256, blk, 0, stream>>>(x1_f, mlp_f, out, (long)M * DM);
}
